// TaggerBiRNN_22093311771304
// MI455X (gfx1250) — matrix operand order repaired and checked by reading, not compiled, not run on hardware
//
#include <hip/hip_runtime.h>
#include <hip/hip_bf16.h>
#include <math.h>

// Problem constants
#define VV 50000
#define EE 300
#define EP 320        // E padded to multiple of 32 (WMMA K)
#define HH 512
#define G3 1536       // 3*H
#define CC 21         // C+1
#define BB 64
#define SS 256
#define MROWS (SS*BB) // 16384

typedef __attribute__((ext_vector_type(16))) __bf16 v16bf;
typedef __attribute__((ext_vector_type(8)))  float  v8f;
typedef __attribute__((ext_vector_type(4)))  unsigned tdm_u4;
typedef __attribute__((ext_vector_type(8)))  int      tdm_i8;
typedef __attribute__((ext_vector_type(4)))  int      tdm_i4;

union Frag16 { uint4 q[2]; v16bf v; };

// ---- WMMA helper (CDNA5 16x16x32 bf16, f32 accum) --------------------------
static __device__ __forceinline__ v8f wmma_bf16(v16bf a, v16bf b, v8f c) {
  return __builtin_amdgcn_wmma_f32_16x16x32_bf16(
      /*neg_a=*/false, a, /*neg_b=*/false, b,
      /*c_mod=*/(short)0, c, /*reuse_a=*/false, /*reuse_b=*/false);
}

// ---- CDNA5 async memory->LDS copy (per-lane 16B, tracked by ASYNCcnt) ------
static __device__ __forceinline__ void async_copy_b128(const void* lds_dst,
                                                       const void* gsrc) {
  unsigned lofs = (unsigned)(size_t)lds_dst;   // low 32 bits = LDS byte offset
  asm volatile("global_load_async_to_lds_b128 %0, %1, off"
               :: "v"(lofs), "v"(gsrc) : "memory");
}
static __device__ __forceinline__ void wait_async0() {
  asm volatile("s_wait_asynccnt 0x0" ::: "memory");
}

// A-matrix 16x32 bf16 fragment from row-major [rows][ld] buffer.
// ISA: lanes0-15 M=0..15 hold K{0..7,16..23}; lanes16-31 hold K{8..15,24..31}.
static __device__ __forceinline__ v16bf load_a_frag(const __hip_bfloat16* base,
                                                    int ld, int row0, int k0, int lane) {
  int m = row0 + (lane & 15);
  int kofs = k0 + ((lane & 16) ? 8 : 0);
  const __hip_bfloat16* p = base + (size_t)m * ld + kofs;
  Frag16 f;
  f.q[0] = *(const uint4*)(p);        // 8 halves
  f.q[1] = *(const uint4*)(p + 16);   // 8 halves, K+16
  return f.v;
}

// B-matrix 32x16 bf16 fragment; memory row-major W[N][K] (computes A*W^T).
// Lanes0-15: col n=lane, K{0..7,16..23} ; lanes16-31: col n=lane-16, K{8..15,24..31}.
static __device__ __forceinline__ v16bf load_b_frag(const __hip_bfloat16* W,
                                                    int ldk, int n0, int k0, int lane) {
  int n = n0 + (lane & 15);
  int kofs = k0 + ((lane & 16) ? 8 : 0);
  const __hip_bfloat16* p = W + (size_t)n * ldk + kofs;
  Frag16 f;
  f.q[0] = *(const uint4*)(p);
  f.q[1] = *(const uint4*)(p + 16);
  return f.v;
}

// ---- Kernel 0: weight conversion / scratch init ----------------------------
__global__ void prep_kernel(const float* Whh_f, const float* Whh_b,
                            const float* Wih_f, const float* Wih_b,
                            __hip_bfloat16* whh_bf,   // [2][1536][512]
                            __hip_bfloat16* wih_bf,   // [3072][320] zero-padded
                            __hip_bfloat16* hbuf,     // [2][2][64][512] -> zero
                            int* bars)                // [2][256] -> zero
{
  const int NW = 2 * G3 * HH;          // 1,572,864 (largest domain)
  for (int i = blockIdx.x * blockDim.x + threadIdx.x; i < NW;
       i += gridDim.x * blockDim.x) {
    {
      int dir = i / (G3 * HH);
      int r   = i % (G3 * HH);
      const float* src = dir ? Whh_b : Whh_f;
      whh_bf[i] = __float2bfloat16(src[r]);
    }
    if (i < 2 * G3 * EP) {
      int row = i / EP, e = i % EP;
      float v = 0.f;
      if (e < EE) v = (row < G3) ? Wih_f[row * EE + e] : Wih_b[(row - G3) * EE + e];
      wih_bf[i] = __float2bfloat16(v);
    }
    if (i < 2 * 2 * BB * HH) hbuf[i] = __float2bfloat16(0.f);
    if (i < 2 * SS) bars[i] = 0;
  }
}

// ---- Kernel 1: embedding gather -> xs bf16 [S*B][EP] -----------------------
__global__ void embed_kernel(const int* tokens, const float* emb,
                             __hip_bfloat16* xs) {
  const int N = MROWS * EP;
  for (int i = blockIdx.x * blockDim.x + threadIdx.x; i < N;
       i += gridDim.x * blockDim.x) {
    int r = i / EP, e = i % EP;
    int s = r / BB, b = r % BB;      // xs row = s*B + b  ([S,B,E] layout)
    float v = 0.f;
    if (e < EE) {
      int tok = tokens[b * SS + s];  // inputs_batch is [B,S]
      v = emb[(size_t)tok * EE + e];
    }
    xs[i] = __float2bfloat16(v);
  }
}

// ---- Kernel 2: gi = xs @ Wih_cat^T + b_ih  (bf16 out, [16384][3072]) -------
// A tile (16 x 320, shared by all 8 waves of the block) staged once into LDS
// via async-to-LDS copies; B streamed from global (L2 resident, 2 MB).
__global__ __launch_bounds__(256)
void gi_gemm_kernel(const __hip_bfloat16* xs, const __hip_bfloat16* wih,
                    const float* bih_f, const float* bih_b,
                    __hip_bfloat16* gi) {
  __shared__ __hip_bfloat16 As[16 * EP];     // 10 KB
  int tid = threadIdx.x, lane = tid & 31, wave = tid >> 5;
  int mt = blockIdx.x;                  // 0..1023
  int nt = blockIdx.y * 8 + wave;       // 0..191

  for (int idx = tid; idx < 16 * EP / 8; idx += blockDim.x) {
    int r = idx / (EP / 8), c8 = (idx % (EP / 8)) * 8;
    async_copy_b128(As + r * EP + c8, xs + (size_t)(mt * 16 + r) * EP + c8);
  }
  wait_async0();
  __syncthreads();

  v8f c = {};
#pragma unroll
  for (int kt = 0; kt < EP / 32; ++kt) {
    v16bf a = load_a_frag(As,  EP, 0,       kt * 32, lane);
    v16bf b = load_b_frag(wih, EP, nt * 16, kt * 32, lane);
    c = wmma_bf16(a, b, c);
  }
  int n = nt * 16 + (lane & 15);
  float bias = (n < G3) ? bih_f[n] : bih_b[n - G3];
  int mbase = mt * 16 + ((lane >> 4) << 3);
#pragma unroll
  for (int v = 0; v < 8; ++v)
    gi[(size_t)(mbase + v) * (2 * G3) + n] = __float2bfloat16(c[v] + bias);
}

// ---- Kernel 3: persistent bidirectional GRU recurrence ---------------------
// 32 WGs: dir = blk/16, wg = blk%16. WG owns hidden units [wg*32, wg*32+32),
// i.e. W_hh rows {j, 512+j, 1024+j} (96x512 bf16) LDS-resident (loaded via the
// Tensor Data Mover where the builtin exists). Per step: async-stage h
// (64x512 bf16) from L2 into LDS, 64x96x512 WMMA GEMM, GRU gates in f32,
// publish h slice, inter-WG barrier.
__global__ __launch_bounds__(256)
void recurrent_kernel(const __hip_bfloat16* whh_bf,    // [2][1536][512]
                      const __hip_bfloat16* gi,        // [16384][3072]
                      const float* bhh_f, const float* bhh_b,
                      __hip_bfloat16* hbuf,            // [2][2][64][512]
                      __hip_bfloat16* hcat,            // [16384][1024]
                      int* bars)                       // [2][256]
{
  extern __shared__ char smem_raw[];
  __hip_bfloat16* Ws   = (__hip_bfloat16*)smem_raw;          // 96*512 bf16 = 96 KB
  __hip_bfloat16* hs   = Ws + 96 * HH;                       // 64*512 bf16 = 64 KB
  float*          ghs  = (float*)(hs + BB * HH);             // [96][64] f32 = 24 KB
  float*          hown = ghs + 96 * BB;                      // [64][32] f32 = 8 KB

  const int dir = blockIdx.x >> 4;
  const int wg  = blockIdx.x & 15;
  const int tid = threadIdx.x, lane = tid & 31, wave = tid >> 5;
  const float* bhh = dir ? bhh_b : bhh_f;
  const __hip_bfloat16* W = whh_bf + (size_t)dir * G3 * HH;

  // ---- One-time load of our 96 W_hh rows (3 gate blocks of 32x512) --------
#if __has_builtin(__builtin_amdgcn_tensor_load_to_lds)
  if (wave == 0) {
#pragma unroll
    for (int g = 0; g < 3; ++g) {
      const __hip_bfloat16* gsrc = W + (size_t)(g * HH + wg * 32) * HH;
      unsigned long long ga = (unsigned long long)(size_t)gsrc;
      unsigned lofs = (unsigned)(size_t)(Ws + g * 32 * HH);
      tdm_u4 g0;
      g0[0] = 1u;                                       // count=1 (valid D#)
      g0[1] = lofs;                                     // lds_addr
      g0[2] = (unsigned)(ga & 0xFFFFFFFFu);             // global_addr[31:0]
      g0[3] = (unsigned)((ga >> 32) & 0x1FFFFFFu) | (2u << 30);  // [56:32] | type=2
      tdm_i8 g1;
      g1[0] = 0x00010000;                               // wg_mask=0, data_size=2B
      g1[1] = (int)((HH & 0xFFFF) << 16);               // tensor_dim0 = 512 (lo16)
      g1[2] = (int)((HH >> 16) | ((32u & 0xFFFFu) << 16)); // dim0 hi | tensor_dim1=32
      g1[3] = (int)((HH & 0xFFFFu) << 16);              // dim1 hi=0 | tile_dim0=512
      g1[4] = 32;                                       // tile_dim1=32, tile_dim2=0
      g1[5] = HH;                                       // tensor_dim0_stride = 512
      g1[6] = 0;
      g1[7] = 0;
      tdm_i4 gz = {0, 0, 0, 0};
#if __clang_major__ >= 23
      tdm_i8 gz8 = {0, 0, 0, 0, 0, 0, 0, 0};
      __builtin_amdgcn_tensor_load_to_lds(g0, g1, gz, gz, gz8, 0);
#else
      __builtin_amdgcn_tensor_load_to_lds(g0, g1, gz, gz, 0);
#endif
    }
    __builtin_amdgcn_s_wait_tensorcnt((short)0);
  }
#else
  for (int idx = tid; idx < 96 * HH / 8; idx += blockDim.x) {
    int j = idx / (HH / 8), c8 = (idx % (HH / 8)) * 8;
    int gr = (j >> 5) * HH + wg * 32 + (j & 31);
    *(uint4*)(Ws + j * HH + c8) = *(const uint4*)(W + (size_t)gr * HH + c8);
  }
#endif
  for (int idx = tid; idx < BB * 32; idx += blockDim.x) hown[idx] = 0.f;
  __syncthreads();

  int pp = 0;
  for (int s = 0; s < SS; ++s) {
    // Async-stage full h state (bf16, 64 KB) into LDS, no VGPR round-trip.
    const __hip_bfloat16* hsrc = hbuf + ((size_t)(dir * 2 + pp)) * BB * HH;
    for (int idx = tid; idx < BB * HH / 8; idx += blockDim.x)
      async_copy_b128(hs + idx * 8, hsrc + idx * 8);
    wait_async0();
    __syncthreads();

    // gh = h @ Wslice^T : M=64 (4 tiles) x N=96 (6 tiles) x K=512 (16 tiles).
#pragma unroll
    for (int t = 0; t < 3; ++t) {
      int tt = wave * 3 + t, mt = tt / 6, nt = tt % 6;
      v8f c = {};
#pragma unroll
      for (int kt = 0; kt < HH / 32; ++kt) {
        v16bf a = load_a_frag(hs, HH, mt * 16, kt * 32, lane);
        v16bf b = load_b_frag(Ws, HH, nt * 16, kt * 32, lane);
        c = wmma_bf16(a, b, c);
      }
      int n = nt * 16 + (lane & 15);
      int mbase = mt * 16 + ((lane >> 4) << 3);
#pragma unroll
      for (int v = 0; v < 8; ++v)              // transposed store: [gateRow][batch]
        ghs[n * BB + mbase + v] = c[v];
    }
    __syncthreads();

    // GRU gates for our 32 units x 64 batch.
    int srow = dir ? (SS - 1 - s) : s;
    for (int idx = tid; idx < 32 * BB; idx += blockDim.x) {
      int j = idx >> 6, b = idx & 63;
      int gj = wg * 32 + j;
      const __hip_bfloat16* gip = gi + (size_t)(srow * BB + b) * (2 * G3) + dir * G3;
      float ir = (float)gip[gj], iz = (float)gip[HH + gj], in_ = (float)gip[2 * HH + gj];
      float hr = ghs[j * BB + b]        + bhh[gj];
      float hz = ghs[(32 + j) * BB + b] + bhh[HH + gj];
      float hn = ghs[(64 + j) * BB + b] + bhh[2 * HH + gj];
      float r = 1.f / (1.f + expf(-(ir + hr)));
      float z = 1.f / (1.f + expf(-(iz + hz)));
      float n = tanhf(in_ + r * hn);
      float hp = hown[b * 32 + j];
      float hnew = (1.f - z) * n + z * hp;
      hown[b * 32 + j] = hnew;
      __hip_bfloat16 h16 = __float2bfloat16(hnew);
      hbuf[((size_t)(dir * 2 + (1 - pp)) * BB + b) * HH + gj] = h16;
      hcat[(size_t)(srow * BB + b) * (2 * HH) + dir * HH + gj] = h16;
    }

    // Inter-WG barrier (16 WGs per direction).
    __threadfence();
    __syncthreads();
    if (tid == 0) {
      int* ctr = bars + dir * SS + s;
      atomicAdd(ctr, 1);
      volatile int* vc = ctr;
      while (*vc < 16) __builtin_amdgcn_s_sleep(2);
    }
    __builtin_amdgcn_s_cluster_barrier();   // NOP unless dispatched as a cluster
    __syncthreads();
    __threadfence();
    pp ^= 1;
  }
}

// ---- Kernel 4: logits + log_softmax, out[B][21][S] -------------------------
__global__ __launch_bounds__(256)
void final_kernel(const __hip_bfloat16* hcat, const float* linW,
                  const float* linb, float* out) {
  int row = blockIdx.x * 8 + (threadIdx.x >> 5);   // s*B + b, one wave per row
  int lane = threadIdx.x & 31;
  int s = row >> 6, b = row & 63;
  const __hip_bfloat16* h = hcat + (size_t)row * (2 * HH);
  float mx = -3.4e38f, se = 0.f, mine = 0.f;
  for (int c = 0; c < CC; ++c) {
    const float* w = linW + c * (2 * HH);
    float p = 0.f;
    for (int k = lane; k < 2 * HH; k += 32) p += (float)h[k] * w[k];
#pragma unroll
    for (int o = 16; o; o >>= 1) p += __shfl_xor(p, o, 32);
    p += linb[c];
    if (lane == c) mine = p;
    float nm = fmaxf(mx, p);                 // online logsumexp
    se = se * expf(mx - nm) + expf(p - nm);
    mx = nm;
  }
  float lse = mx + logf(se);
  if (lane < CC)
    out[((size_t)b * CC + lane) * SS + s] = mine - lse;
}

// ---- Host launcher ---------------------------------------------------------
extern "C" void kernel_launch(void* const* d_in, const int* in_sizes, int n_in,
                              void* d_out, int out_size, void* d_ws, size_t ws_size,
                              hipStream_t stream) {
  const int*   tokens = (const int*)  d_in[0];
  const float* emb    = (const float*)d_in[1];
  const float* Wih_f  = (const float*)d_in[2];
  const float* Whh_f  = (const float*)d_in[3];
  const float* bih_f  = (const float*)d_in[4];
  const float* bhh_f  = (const float*)d_in[5];
  const float* Wih_b  = (const float*)d_in[6];
  const float* Whh_b  = (const float*)d_in[7];
  const float* bih_b  = (const float*)d_in[8];
  const float* bhh_b  = (const float*)d_in[9];
  const float* linW   = (const float*)d_in[10];
  const float* linb   = (const float*)d_in[11];
  float* out = (float*)d_out;

  // Workspace layout (bytes)
  char* ws = (char*)d_ws;
  size_t o_whh  = 0;                                     // 2*1536*512 bf16
  size_t o_wih  = o_whh  + (size_t)2 * G3 * HH * 2;      // 3072*320 bf16
  size_t o_xs   = o_wih  + (size_t)2 * G3 * EP * 2;      // 16384*320 bf16
  size_t o_gi   = o_xs   + (size_t)MROWS * EP * 2;       // 16384*3072 bf16
  size_t o_hcat = o_gi   + (size_t)MROWS * 2 * G3 * 2;   // 16384*1024 bf16
  size_t o_hbuf = o_hcat + (size_t)MROWS * 2 * HH * 2;   // 2*2*64*512 bf16
  size_t o_bar  = o_hbuf + (size_t)2 * 2 * BB * HH * 2;  // 2*256 int

  __hip_bfloat16* whh_bf = (__hip_bfloat16*)(ws + o_whh);
  __hip_bfloat16* wih_bf = (__hip_bfloat16*)(ws + o_wih);
  __hip_bfloat16* xs     = (__hip_bfloat16*)(ws + o_xs);
  __hip_bfloat16* gi     = (__hip_bfloat16*)(ws + o_gi);
  __hip_bfloat16* hcat   = (__hip_bfloat16*)(ws + o_hcat);
  __hip_bfloat16* hbuf   = (__hip_bfloat16*)(ws + o_hbuf);
  int*            bars   = (int*)           (ws + o_bar);

  const size_t rec_lds = (96 * HH + BB * HH) * sizeof(__hip_bfloat16)
                       + (96 * BB + BB * 32) * sizeof(float);   // 192 KB
  (void)hipFuncSetAttribute((const void*)recurrent_kernel,
                            hipFuncAttributeMaxDynamicSharedMemorySize,
                            (int)rec_lds);

  prep_kernel<<<6144, 256, 0, stream>>>(Whh_f, Whh_b, Wih_f, Wih_b,
                                        whh_bf, wih_bf, hbuf, bars);
  embed_kernel<<<20480, 256, 0, stream>>>(tokens, emb, xs);
  gi_gemm_kernel<<<dim3(MROWS / 16, (2 * G3) / (16 * 8)), 256, 0, stream>>>(
      xs, wih_bf, bih_f, bih_b, gi);
  recurrent_kernel<<<32, 256, rec_lds, stream>>>(whh_bf, gi, bhh_f, bhh_b,
                                                 hbuf, hcat, bars);
  final_kernel<<<MROWS / 8, 256, 0, stream>>>(hcat, linW, linb, out);
}
